// GambaMulti_37855841747199
// MI455X (gfx1250) — compile-verified, matrix operand order checked by reading, not run on hardware
//
// Gamba (GNN + Mamba) forward for MI455X / gfx1250, compile-only build.
//
// All dense GEMMs via v_wmma_f32_16x16x32_bf16 (bf16 A/B, f32 acc): bf16 WMMA
// has 8x the K-depth per instruction vs the f32 16x16x4 path, weights are
// 0.02*N(0,1) and every MLP is layernormed, so bf16 storage + f32 accumulate
// is the right operating point. Activations (32-96MB) are L2-resident (192MB).
//
// Data movement: A row-blocks are staged to LDS with global_load_async_to_lds_b128
// (ASYNCcnt + s_wait_asynccnt, CDNA5 ISA), fragments are pure
// b128 loads (no per-element guards: theta padded 4->16 rows, dt_w padded
// K 16->32 with zero rows, gated weights pre-transposed at staging).
//
// Param pytree assumed flattened in JAX order (dicts sorted by key):
// dec(2x{b,ln_b,ln_w,w}) enc(2x) gated{b_hh,b_ih,w_hh,w_ih,weight}
// mamba 3x{blocks 4x{A_log,D,conv_b,conv_w,dt_b,dt_w,in_proj,norm,out_proj,x_proj}, norm_f}
// merge 3x1x{b,ln_b,ln_w,w} out_gin{b,w} theta 3x  => 161 leaves, n_in = 164.

#include <hip/hip_runtime.h>

typedef __bf16 bf16;
typedef __attribute__((ext_vector_type(16))) __bf16 v16bf;
typedef __attribute__((ext_vector_type(8)))  float  v8f;

#define ACT_NONE     0
#define ACT_RELU     1
#define ACT_SOFTPLUS 2

#define NN      65536
#define NGRAPH  128
#define PP      512
#define HH      128

#define USE_ASYNC_LDS 1

union FragU { v16bf v; bf16 e[16]; uint4 q[2]; };

__device__ __forceinline__ bf16 f2bf(float f) {
  union { float f; unsigned int u; } a; a.f = f;
  unsigned int r = a.u + 0x7FFFu + ((a.u >> 16) & 1u);
  union { unsigned short s; bf16 b; } o; o.s = (unsigned short)(r >> 16);
  return o.b;
}
__device__ __forceinline__ bf16 bfzero() {
  union { unsigned short s; bf16 b; } o; o.s = 0; return o.b;
}

// 16-byte chunk: global memory -> LDS, asynchronous (tracked by ASYNCcnt).
__device__ __forceinline__ void async_ld16(bf16* lds, const bf16* gaddr) {
#if USE_ASYNC_LDS
  unsigned l = (unsigned)(unsigned long long)lds;   // low 32 bits = LDS offset
  asm volatile("global_load_async_to_lds_b128 %0, %1, off"
               :: "v"(l), "v"(gaddr) : "memory");
#else
  *(uint4*)lds = *(const uint4*)gaddr;
#endif
}
__device__ __forceinline__ void wait_async() {
#if USE_ASYNC_LDS
  asm volatile("s_wait_asynccnt 0x0" ::: "memory");
#endif
}

// K-index mapping for a 16-bit fragment: lanes 0-15 hold K 0-7 & 16-23,
// lanes 16-31 hold K 8-15 & 24-31 (ISA 7.12.2, 16-bit A 16x32 layout).
__device__ __forceinline__ int frag_kk(int e, int half) {
  return (e < 8) ? (half * 8 + e) : (16 + half * 8 + (e - 8));
}

// Branch-free fragment from 16 row-major rows of `ld` bf16 elements
// (base pre-offset to the first row). Two b128 loads per lane.
// Works for A tiles in LDS (ld = Kpad) and W[N,K] tiles in global memory.
__device__ __forceinline__ v16bf frag_rows16(const bf16* base, long ld, int k0, int lane) {
  const bf16* row = base + (long)(lane & 15) * ld + k0 + ((lane >> 4) & 1) * 8;
  FragU f;
  f.q[0] = *(const uint4*)row;
  f.q[1] = *(const uint4*)(row + 16);
  return f.v;
}

// Branch-free strided gather for non-transposed B[K,N] (only the batched
// alpha @ x_dense GEMM uses this; K multiple of 32 and all cols valid there).
__device__ __forceinline__ v16bf frag_b_gather(const bf16* B, long ld, int c0, int k0, int lane) {
  int c = c0 + (lane & 15);
  int half = (lane >> 4) & 1;
  FragU f;
#pragma unroll
  for (int e = 0; e < 16; ++e)
    f.e[e] = B[(long)(k0 + frag_kk(e, half)) * ld + c];
  return f.v;
}

__device__ __forceinline__ float act_apply(float v, int act) {
  if (act == ACT_RELU) return fmaxf(v, 0.0f);
  if (act == ACT_SOFTPLUS) return (v > 20.0f) ? v : log1pf(expf(v));
  return v;
}

// ---------------------------------------------------------------------------
// Generic WMMA GEMM: C[M,N] = epilogue(A[M,K] * op(B) + bias (+resid)).
// bT=1: B is W[Nb,K] row-major, Nb >= ceil16(N) rows exist (padded).
// bT=0: B is [K,N] row-major (all tile columns valid).
// A 16-row block is async-staged to LDS once per WG; waves own N-tiles.
// doLN: LayerNorm over N (N<=128) staged in LDS, then act.
// Cbf: optional dual bf16 output. blockIdx.z batches with given strides.
// ---------------------------------------------------------------------------
__global__ __launch_bounds__(256) void k_gemm(
    const bf16* __restrict__ A, long lda, long sA_,
    const bf16* __restrict__ B, long ldb, long sB_, int bT,
    const float* __restrict__ bias,
    const float* __restrict__ resid, long ldr, long sR_,
    float* __restrict__ C, long ldc, long sC_,
    bf16* __restrict__ Cbf,
    int M, int N, int K,
    int act, int doLN,
    const float* __restrict__ lnw, const float* __restrict__ lnb)
{
  __shared__ bf16 sA[16 * 1024];            // staged A row-block (K <= 1024)
  __shared__ float cbuf[16 * 128];
  __shared__ float s_mu[16], s_rs[16];

  int z = blockIdx.z;
  const bf16* Ab = A + (long)z * sA_;
  const bf16* Bb = B + (long)z * sB_;
  float* Cb = C + (long)z * sC_;
  const float* Rb = resid ? (resid + (long)z * sR_) : (const float*)0;
  bf16* Cbb = Cbf ? (Cbf + (long)z * sC_) : (bf16*)0;

  int r0 = blockIdx.x * 16;
  int lane = threadIdx.x & 31;
  int wave = threadIdx.x >> 5;
  int Kpad = (K + 31) & ~31;
  int nch = K >> 3;                          // 16B chunks per row (K % 8 == 0)

  // ---- async stage A[r0:r0+16, 0:K] into LDS ----
  for (int c = threadIdx.x; c < 16 * nch; c += 256) {
    int rr = c / nch, cc = c - rr * nch;
    int row = r0 + rr;
    if (row < M)
      async_ld16(&sA[rr * Kpad + cc * 8], Ab + (long)row * lda + cc * 8);
  }
  wait_async();
  __syncthreads();

  int nT = (N + 15) >> 4;
  for (int tt = wave; tt < nT; tt += 8) {
    int c0 = tt << 4;
    const bf16* Brow = Bb + (bT ? (long)c0 * ldb : 0);
    v8f acc = {};
    for (int k0 = 0; k0 < K; k0 += 32) {
      __builtin_prefetch(Brow + (long)(lane & 15) * ldb + k0 + 32, 0, 1);
      v16bf a = frag_rows16(sA, Kpad, k0, lane);
      v16bf b = bT ? frag_rows16(Brow, ldb, k0, lane)
                   : frag_b_gather(Bb, ldb, c0, k0, lane);
      acc = __builtin_amdgcn_wmma_f32_16x16x32_bf16(
          false, a, false, b, (short)0, acc, false, false);
    }
    int col = c0 + (lane & 15);
    int half = lane >> 4;
    if (col < N) {
      float bv = bias ? bias[col] : 0.0f;
#pragma unroll
      for (int r = 0; r < 8; ++r) {
        int rr = half * 8 + r;               // row within 16-row block (C layout)
        int row = r0 + rr;
        if (row < M) {
          float v = acc[r] + bv;
          if (Rb) v += Rb[(long)row * ldr + col];
          if (doLN) {
            cbuf[rr * N + col] = v;          // N <= 128 guaranteed when doLN
          } else {
            v = act_apply(v, act);
            Cb[(long)row * ldc + col] = v;
            if (Cbb) Cbb[(long)row * ldc + col] = f2bf(v);
          }
        }
      }
    }
  }

  if (doLN) {
    __syncthreads();
    if (threadIdx.x < 16) {
      int rr = (int)threadIdx.x;
      float mu = 0.0f;
      for (int j = 0; j < N; ++j) mu += cbuf[rr * N + j];
      mu /= (float)N;
      float var = 0.0f;
      for (int j = 0; j < N; ++j) { float d = cbuf[rr * N + j] - mu; var += d * d; }
      var /= (float)N;
      s_mu[rr] = mu;
      s_rs[rr] = rsqrtf(var + 1e-5f);
    }
    __syncthreads();
    for (int idx = threadIdx.x; idx < 16 * N; idx += 256) {
      int rr = idx / N, j = idx - rr * N;
      int row = r0 + rr;
      if (row < M) {
        float v = (cbuf[idx] - s_mu[rr]) * s_rs[rr] * lnw[j] + lnb[j];
        v = act_apply(v, act);
        Cb[(long)row * ldc + j] = v;
        if (Cbb) Cbb[(long)row * ldc + j] = f2bf(v);
      }
    }
  }
}

// ---------------------------------------------------------------------------
// Fused GatedGraphConv GRU step: gi = m*W_ih^T + b_ih, gh = h*W_hh^T + b_hh
// as 48 WMMA tiles into LDS (A tiles async-staged), then gate math; h updated
// in place (each WG touches only its own 16 rows). Never materializes the
// two [65536,384] gate tensors.
// ---------------------------------------------------------------------------
__global__ __launch_bounds__(256) void k_gru(
    const bf16* __restrict__ m_bf, const bf16* __restrict__ h_bf_in,
    const bf16* __restrict__ w_ih, const bf16* __restrict__ w_hh,
    const float* __restrict__ b_ih, const float* __restrict__ b_hh,
    float* __restrict__ h, bf16* __restrict__ h_bf_out, int M)
{
  __shared__ bf16 sM[16 * 128];
  __shared__ bf16 sH[16 * 128];
  __shared__ float gi_s[16 * 384];
  __shared__ float gh_s[16 * 384];

  int r0 = blockIdx.x * 16;
  int lane = threadIdx.x & 31;
  int wave = threadIdx.x >> 5;

  // stage both A tiles: 16 rows x 16 chunks each; 256 threads = 256 chunks
  {
    int rr = threadIdx.x >> 4, cc = threadIdx.x & 15;
    long off = (long)(r0 + rr) * 128 + cc * 8;
    async_ld16(&sM[rr * 128 + cc * 8], m_bf + off);
    async_ld16(&sH[rr * 128 + cc * 8], h_bf_in + off);
  }
  wait_async();
  __syncthreads();

  for (int tt = wave; tt < 48; tt += 8) {
    bool isI = tt < 24;
    int c0 = (isI ? tt : tt - 24) << 4;
    const bf16* Ap = isI ? sM : sH;
    const bf16* Wrow = (isI ? w_ih : w_hh) + (long)c0 * 128;
    v8f acc = {};
    for (int k0 = 0; k0 < 128; k0 += 32) {
      v16bf a = frag_rows16(Ap, 128, k0, lane);
      v16bf b = frag_rows16(Wrow, 128, k0, lane);
      acc = __builtin_amdgcn_wmma_f32_16x16x32_bf16(
          false, a, false, b, (short)0, acc, false, false);
    }
    float* Dst = isI ? gi_s : gh_s;
    int col = c0 + (lane & 15);
    int half = lane >> 4;
#pragma unroll
    for (int r = 0; r < 8; ++r) Dst[(half * 8 + r) * 384 + col] = acc[r];
  }
  __syncthreads();

  for (int idx = threadIdx.x; idx < 16 * 128; idx += 256) {
    int rr = idx >> 7, j = idx & 127;
    int row = r0 + rr;
    if (row >= M) continue;
    float ir = gi_s[rr * 384 + j]       + b_ih[j];
    float iz = gi_s[rr * 384 + 128 + j] + b_ih[128 + j];
    float ig = gi_s[rr * 384 + 256 + j] + b_ih[256 + j];
    float hr = gh_s[rr * 384 + j]       + b_hh[j];
    float hz = gh_s[rr * 384 + 128 + j] + b_hh[128 + j];
    float hg = gh_s[rr * 384 + 256 + j] + b_hh[256 + j];
    float rg = 1.0f / (1.0f + expf(-(ir + hr)));
    float zg = 1.0f / (1.0f + expf(-(iz + hz)));
    float gg = tanhf(ig + rg * hg);
    float hv = h[(long)row * 128 + j];
    float hn = (1.0f - zg) * gg + zg * hv;
    h[(long)row * 128 + j] = hn;
    h_bf_out[(long)row * 128 + j] = f2bf(hn);
  }
}

// ----------------------------- small kernels -------------------------------

__global__ void k_f32_bf16(const float* __restrict__ a, bf16* __restrict__ o, long n) {
  long i = (long)blockIdx.x * blockDim.x + threadIdx.x;
  for (; i < n; i += (long)gridDim.x * blockDim.x) o[i] = f2bf(a[i]);
}

// f32 [R,C] -> bf16 [Rp,Cp], zero padded (for theta rows, dt_w K padding)
__global__ void k_cvt_pad(const float* __restrict__ src, bf16* __restrict__ dst,
                          int R, int C, int Rp, int Cp) {
  long idx = (long)blockIdx.x * blockDim.x + threadIdx.x;
  if (idx >= (long)Rp * Cp) return;
  int r = (int)(idx / Cp), c = (int)(idx - (long)r * Cp);
  dst[idx] = (r < R && c < C) ? f2bf(src[(long)r * C + c]) : bfzero();
}

// f32 [K,N] -> bf16 [N,K] transpose (gated conv weights -> W[N,K] layout)
__global__ void k_cvt_T(const float* __restrict__ src, bf16* __restrict__ dst,
                        int K, int N) {
  long idx = (long)blockIdx.x * blockDim.x + threadIdx.x;
  if (idx >= (long)N * K) return;
  int n = (int)(idx / K), k = (int)(idx - (long)n * K);
  dst[idx] = f2bf(src[(long)k * N + n]);
}

__global__ void k_zero(float* __restrict__ p, long n) {
  long i = (long)blockIdx.x * blockDim.x + threadIdx.x;
  for (; i < n; i += (long)gridDim.x * blockDim.x) p[i] = 0.0f;
}

// wave per edge, 128 f32 cols: out[dst] += vals[src]
__global__ void k_edge_scatter(const float* __restrict__ vals,
                               const int* __restrict__ src, const int* __restrict__ dst,
                               float* __restrict__ out, int nE) {
  long t = (long)blockIdx.x * blockDim.x + threadIdx.x;
  int e = (int)(t >> 5), lane = (int)(t & 31);
  if (e >= nE) return;
  int s = src[e], d = dst[e];
#pragma unroll
  for (int j = 0; j < 4; ++j) {
    int c = lane + j * 32;
    atomicAdd(&out[(long)d * 128 + c], vals[(long)s * 128 + c]);
  }
}

// xc = concat(x, pe) -> bf16 [N,256]
__global__ void k_concat_xc(const float* __restrict__ x, const float* __restrict__ pe,
                            bf16* __restrict__ o, int n) {
  long idx = (long)blockIdx.x * blockDim.x + threadIdx.x;
  if (idx >= (long)n * 256) return;
  long row = idx >> 8; int c = (int)(idx & 255);
  float v = (c < 128) ? x[row * 128 + c] : pe[row * 128 + (c - 128)];
  o[idx] = f2bf(v);
}

// alpha_pre [N,4] f32 -> alpha_t [G,4,512] bf16 (the (0,2,1) transpose)
__global__ void k_alpha_T(const float* __restrict__ ap, bf16* __restrict__ at, int n) {
  long idx = (long)blockIdx.x * blockDim.x + threadIdx.x;
  if (idx >= (long)n * 4) return;
  long nd = idx >> 2; int v = (int)(idx & 3);
  long g = nd >> 9, p = nd & 511;
  at[((long)g * 4 + v) * 512 + p] = f2bf(ap[idx]);
}

__global__ void k_rmsnorm(const float* __restrict__ x, const float* __restrict__ w,
                          bf16* __restrict__ ob, float* __restrict__ of,
                          int rows, int cols) {
  int r = blockIdx.x * blockDim.x + threadIdx.x;
  if (r >= rows) return;
  const float* xr = x + (long)r * cols;
  float ss = 0.0f;
  for (int j = 0; j < cols; ++j) ss += xr[j] * xr[j];
  float sc = rsqrtf(ss / (float)cols + 1e-5f);
  for (int j = 0; j < cols; ++j) {
    float v = xr[j] * sc * w[j];
    if (ob) ob[(long)r * cols + j] = f2bf(v);
    if (of) of[(long)r * cols + j] = v;
  }
}

// causal depthwise conv (K=4) over L=4 + SiLU; xs from in_proj cols [0,512)
__global__ void k_conv_silu(const float* __restrict__ ip, const float* __restrict__ cw,
                            const float* __restrict__ cb, float* __restrict__ xs,
                            bf16* __restrict__ xs_bf) {
  long idx = (long)blockIdx.x * blockDim.x + threadIdx.x;
  if (idx >= (long)NGRAPH * 4 * 512) return;
  int i = (int)(idx & 511);
  long t = idx >> 9;
  int l = (int)(t & 3);
  long g = t >> 2;
  float acc = cb[i];
#pragma unroll
  for (int k = 0; k < 4; ++k) {
    int ls = l - 3 + k;
    if (ls >= 0) acc += ip[((g * 4 + ls) * 1024) + i] * cw[i * 4 + k];
  }
  float v = acc / (1.0f + expf(-acc));   // SiLU
  xs[idx] = v;
  xs_bf[idx] = f2bf(v);
}

// selective scan over L=4, state 16, fused D-skip and SiLU(gate) multiply
__global__ void k_scan(const float* __restrict__ dt, const float* __restrict__ ssm,
                       const float* __restrict__ xs, const float* __restrict__ ip,
                       const float* __restrict__ A_log, const float* __restrict__ Dp,
                       bf16* __restrict__ y_bf) {
  long idx = (long)blockIdx.x * blockDim.x + threadIdx.x;
  if (idx >= (long)NGRAPH * 512) return;
  long g = idx >> 9;
  int i = (int)(idx & 511);
  float A[16], s[16];
#pragma unroll
  for (int n = 0; n < 16; ++n) { A[n] = -expf(A_log[i * 16 + n]); s[n] = 0.0f; }
  float Dv = Dp[i];
  for (int l = 0; l < 4; ++l) {
    long t = g * 4 + l;
    float dtv = dt[t * 512 + i];
    float xv = xs[t * 512 + i];
    float yv = 0.0f;
#pragma unroll
    for (int n = 0; n < 16; ++n) {
      float dA = expf(dtv * A[n]);
      float dBu = dtv * ssm[t * 48 + 16 + n] * xv;
      s[n] = dA * s[n] + dBu;
      yv += s[n] * ssm[t * 48 + 32 + n];
    }
    yv += xv * Dv;
    float gt = ip[t * 1024 + 512 + i];
    yv *= gt / (1.0f + expf(-gt));       // * SiLU(gate)
    y_bf[t * 512 + i] = f2bf(yv);
  }
}

// merge input: concat(x_orig[N,128], mamba last token via batch) -> bf16 [N,384]
__global__ void k_take_last_concat(const float* __restrict__ x_orig,
                                   const float* __restrict__ xm,
                                   const int* __restrict__ batch,
                                   bf16* __restrict__ o, int n) {
  long idx = (long)blockIdx.x * blockDim.x + threadIdx.x;
  if (idx >= (long)n * 384) return;
  long row = idx / 384;
  int c = (int)(idx - row * 384);
  float v;
  if (c < 128) v = x_orig[row * 128 + c];
  else         v = xm[((long)batch[row] * 4 + 3) * 256 + (c - 128)];
  o[idx] = f2bf(v);
}

__global__ void k_add2_bf16(const float* __restrict__ a, const float* __restrict__ b,
                            bf16* __restrict__ o, long n) {
  long i = (long)blockIdx.x * blockDim.x + threadIdx.x;
  for (; i < n; i += (long)gridDim.x * blockDim.x) o[i] = f2bf(a[i] + b[i]);
}

// global_add_pool over the 512 nodes of each graph
__global__ void k_pool(const float* __restrict__ x, float* __restrict__ o,
                       bf16* __restrict__ ob) {
  int idx = blockIdx.x * blockDim.x + threadIdx.x;
  if (idx >= NGRAPH * HH) return;
  int g = idx >> 7, c = idx & 127;
  float s = 0.0f;
  for (int p = 0; p < PP; ++p) s += x[((long)g * PP + p) * HH + c];
  o[idx] = s;
  ob[idx] = f2bf(s);
}

// ------------------------------- host side ---------------------------------

static void gemm(hipStream_t st,
                 const bf16* A, long lda, long sA,
                 const bf16* B, long ldb, long sB, int bT,
                 const float* bias, const float* resid, long ldr, long sR,
                 float* C, long ldc, long sC, bf16* Cbf,
                 int M, int N, int K, int act, int doLN,
                 const float* lnw, const float* lnb, int batches) {
  dim3 g((unsigned)((M + 15) / 16), 1, (unsigned)batches);
  k_gemm<<<g, dim3(256), 0, st>>>(A, lda, sA, B, ldb, sB, bT, bias, resid, ldr, sR,
                                  C, ldc, sC, Cbf, M, N, K, act, doLN, lnw, lnb);
}

extern "C" void kernel_launch(void* const* d_in, const int* in_sizes, int n_in,
                              void* d_out, int out_size, void* d_ws, size_t ws_size,
                              hipStream_t stream) {
  (void)out_size; (void)ws_size;
  if (n_in < 164) return;

  const float* xin  = (const float*)d_in[0];
  const int*   eix  = (const int*)d_in[1];
  const int*   batch = (const int*)d_in[2];
  int nE = in_sizes[1] / 2;
  const int* esrc = eix;
  const int* edst = eix + nE;

  const float* prm[161];
  for (int i = 0; i < 161; ++i) prm[i] = (const float*)d_in[3 + i];
  auto mb = [](int i, int j) { return 21 + i * 41 + j * 10; };
  const int DEC0_B = 0, DEC0_LNB = 1, DEC0_LNW = 2, DEC0_W = 3;
  const int DEC1_B = 4, DEC1_LNB = 5, DEC1_LNW = 6, DEC1_W = 7;
  const int ENC0_B = 8, ENC0_LNB = 9, ENC0_LNW = 10, ENC0_W = 11;
  const int ENC1_B = 12, ENC1_LNB = 13, ENC1_LNW = 14, ENC1_W = 15;
  const int G_BHH = 16, G_BIH = 17, G_WHH = 18, G_WIH = 19, G_WT = 20;
  const int MRG0 = 144, OG_B = 156, OG_W = 157, TH0 = 158;

  size_t off = 0;
  auto alloc = [&](size_t bytes) -> void* {
    void* p = (char*)d_ws + off;
    off += (bytes + 255) & ~(size_t)255;
    return p;
  };
  auto cvt = [&](const float* s, bf16* d, long n) {
    unsigned b = (unsigned)((n + 255) / 256);
    if (b > 65535u) b = 65535u;
    k_f32_bf16<<<dim3(b), dim3(256), 0, stream>>>(s, d, n);
  };
  auto cvtpad = [&](const float* s, bf16* d, int R, int C, int Rp, int Cp) {
    long n = (long)Rp * Cp;
    k_cvt_pad<<<dim3((unsigned)((n + 255) / 256)), dim3(256), 0, stream>>>(s, d, R, C, Rp, Cp);
  };

  // ---- bf16 weight staging (padded / transposed where needed) ----
  bf16* wb_enc0 = (bf16*)alloc(128 * 64 * 2);   cvt(prm[ENC0_W], wb_enc0, 128 * 64);
  bf16* wb_enc1 = (bf16*)alloc(128 * 128 * 2);  cvt(prm[ENC1_W], wb_enc1, 128 * 128);
  bf16* wb_gw   = (bf16*)alloc(8 * 128 * 128 * 2);
  for (int l = 0; l < 8; ++l)   // pre-transpose [K,N] -> [N,K] so hW uses bT=1
    k_cvt_T<<<dim3((128 * 128 + 255) / 256), dim3(256), 0, stream>>>(
        prm[G_WT] + (long)l * 128 * 128, wb_gw + (long)l * 128 * 128, 128, 128);
  bf16* wb_wih  = (bf16*)alloc(384 * 128 * 2);  cvt(prm[G_WIH], wb_wih, 384 * 128);
  bf16* wb_whh  = (bf16*)alloc(384 * 128 * 2);  cvt(prm[G_WHH], wb_whh, 384 * 128);
  bf16* wb_og   = (bf16*)alloc(128 * 128 * 2);  cvt(prm[OG_W], wb_og, 128 * 128);
  bf16* wb_dec0 = (bf16*)alloc(128 * 128 * 2);  cvt(prm[DEC0_W], wb_dec0, 128 * 128);
  bf16* wb_dec1 = (bf16*)alloc(32 * 128 * 2);   cvt(prm[DEC1_W], wb_dec1, 32 * 128);
  bf16* wb_th[3]; bf16* wb_mrg[3];
  bf16 *wb_ip[3][4], *wb_xp[3][4], *wb_dt[3][4], *wb_op[3][4];
  for (int i = 0; i < 3; ++i) {
    wb_th[i]  = (bf16*)alloc(16 * 256 * 2);               // theta padded 4 -> 16 rows
    cvtpad(prm[TH0 + i], wb_th[i], 4, 256, 16, 256);
    wb_mrg[i] = (bf16*)alloc(128 * 384 * 2);  cvt(prm[MRG0 + i * 4 + 3], wb_mrg[i], 128 * 384);
    for (int j = 0; j < 4; ++j) {
      int b = mb(i, j);
      wb_ip[i][j] = (bf16*)alloc(1024 * 256 * 2); cvt(prm[b + 6], wb_ip[i][j], 1024 * 256);
      wb_xp[i][j] = (bf16*)alloc(48 * 512 * 2);   cvt(prm[b + 9], wb_xp[i][j], 48 * 512);
      wb_dt[i][j] = (bf16*)alloc(512 * 32 * 2);             // dt_w padded K 16 -> 32
      cvtpad(prm[b + 5], wb_dt[i][j], 512, 16, 512, 32);
      wb_op[i][j] = (bf16*)alloc(256 * 512 * 2);  cvt(prm[b + 8], wb_op[i][j], 256 * 512);
    }
  }

  // ---- activation scratch ----
  float* x_orig = (float*)alloc((size_t)NN * 128 * 4);
  float* x_cur  = (float*)alloc((size_t)NN * 128 * 4);
  float* h      = (float*)alloc((size_t)NN * 128 * 4);
  float* bufF1  = (float*)alloc((size_t)NN * 128 * 4);
  float* msum   = (float*)alloc((size_t)NN * 128 * 4);
  bf16*  h_bf   = (bf16*)alloc((size_t)NN * 128 * 2);
  bf16*  bufB1  = (bf16*)alloc((size_t)NN * 128 * 2);
  bf16*  abuf   = (bf16*)alloc((size_t)NN * 384 * 2);
  bf16*  xin_bf = (bf16*)alloc((size_t)NN * 64 * 2);
  float* alpha_pre = (float*)alloc((size_t)NN * 4 * 4);
  bf16*  alpha_t   = (bf16*)alloc((size_t)NGRAPH * 4 * PP * 2);
  float* hm   = (float*)alloc(512 * 256 * 4);
  bf16*  rn_bf = (bf16*)alloc(512 * 256 * 2);
  float* ip   = (float*)alloc(512 * 1024 * 4);
  float* xs   = (float*)alloc(512 * 512 * 4);
  bf16*  xs_bf = (bf16*)alloc(512 * 512 * 2);
  float* ssm  = (float*)alloc(512 * 48 * 4);
  bf16*  ssm_bf = (bf16*)alloc(512 * 48 * 2);
  float* dtb  = (float*)alloc(512 * 512 * 4);
  bf16*  y_bf = (bf16*)alloc(512 * 512 * 2);
  float* xmf  = (float*)alloc(512 * 256 * 4);
  float* poolf = (float*)alloc(128 * 128 * 4);
  bf16*  pool_bf = (bf16*)alloc(128 * 128 * 2);
  float* ddf  = (float*)alloc(128 * 128 * 4);
  bf16*  dd_bf = (bf16*)alloc(128 * 128 * 2);

  const long n128 = (long)NN * 128;

  // ---- encoder MLP ----
  cvt(xin, xin_bf, (long)NN * 64);
  gemm(stream, xin_bf, 64, 0, wb_enc0, 64, 0, 1, prm[ENC0_B], 0, 0, 0,
       bufF1, 128, 0, bufB1, NN, 128, 64, ACT_RELU, 1, prm[ENC0_LNW], prm[ENC0_LNB], 1);
  gemm(stream, bufB1, 128, 0, wb_enc1, 128, 0, 1, prm[ENC1_B], 0, 0, 0,
       x_orig, 128, 0, h_bf, NN, 128, 128, ACT_NONE, 1, prm[ENC1_LNW], prm[ENC1_LNB], 1);
  hipMemcpyAsync(h, x_orig, (size_t)n128 * 4, hipMemcpyDeviceToDevice, stream);
  hipMemcpyAsync(x_cur, x_orig, (size_t)n128 * 4, hipMemcpyDeviceToDevice, stream);

  // ---- GatedGraphConv: 8 steps ----
  for (int l = 0; l < 8; ++l) {
    gemm(stream, h_bf, 128, 0, wb_gw + (long)l * 128 * 128, 128, 0, 1, 0, 0, 0, 0,
         bufF1, 128, 0, 0, NN, 128, 128, ACT_NONE, 0, 0, 0, 1);
    k_zero<<<dim3(8192), dim3(256), 0, stream>>>(msum, n128);
    k_edge_scatter<<<dim3((unsigned)(((long)nE * 32 + 255) / 256)), dim3(256), 0, stream>>>(
        bufF1, esrc, edst, msum, nE);
    cvt(msum, bufB1, n128);
    k_gru<<<dim3(NN / 16), dim3(256), 0, stream>>>(
        bufB1, h_bf, wb_wih, wb_whh, prm[G_BIH], prm[G_BHH], h, h_bf, NN);
  }
  // pe == h from here on

  // ---- 3 Gamba layers ----
  for (int i = 0; i < 3; ++i) {
    k_concat_xc<<<dim3((unsigned)(((long)NN * 256 + 255) / 256)), dim3(256), 0, stream>>>(
        x_cur, h, abuf, NN);                                // abuf = xc bf16 [N,256]
    gemm(stream, abuf, 256, 0, wb_th[i], 256, 0, 1, 0, 0, 0, 0,
         alpha_pre, 4, 0, 0, NN, 4, 256, ACT_NONE, 0, 0, 0, 1);
    k_alpha_T<<<dim3((unsigned)(((long)NN * 4 + 255) / 256)), dim3(256), 0, stream>>>(
        alpha_pre, alpha_t, NN);
    // alpha_x[g] = alpha_t[g] (4x512) @ xc_dense[g] (512x256), batched over graphs
    gemm(stream, alpha_t, 512, 4 * 512, abuf, 256, (long)PP * 256, 0, 0, 0, 0, 0,
         hm, 256, 4 * 256, 0, 4, 256, 512, ACT_NONE, 0, 0, 0, NGRAPH);

    for (int j = 0; j < 4; ++j) {
      int b = mb(i, j);
      k_rmsnorm<<<dim3(2), dim3(256), 0, stream>>>(hm, prm[b + 7], rn_bf, (float*)0, 512, 256);
      gemm(stream, rn_bf, 256, 0, wb_ip[i][j], 256, 0, 1, 0, 0, 0, 0,
           ip, 1024, 0, 0, 512, 1024, 256, ACT_NONE, 0, 0, 0, 1);
      k_conv_silu<<<dim3(1024), dim3(256), 0, stream>>>(ip, prm[b + 3], prm[b + 2], xs, xs_bf);
      gemm(stream, xs_bf, 512, 0, wb_xp[i][j], 512, 0, 1, 0, 0, 0, 0,
           ssm, 48, 0, ssm_bf, 512, 48, 512, ACT_NONE, 0, 0, 0, 1);
      // dt: K padded to 32 (wb_dt rows 16..31 are zero; A cols 16..31 of the
      // 48-wide ssm row multiply the zero rows, so the pad is inert)
      gemm(stream, ssm_bf, 48, 0, wb_dt[i][j], 32, 0, 1, prm[b + 4], 0, 0, 0,
           dtb, 512, 0, 0, 512, 512, 32, ACT_SOFTPLUS, 0, 0, 0, 1);
      k_scan<<<dim3(256), dim3(256), 0, stream>>>(dtb, ssm, xs, ip, prm[b + 0], prm[b + 1], y_bf);
      gemm(stream, y_bf, 512, 0, wb_op[i][j], 512, 0, 1, 0, hm, 256, 0,
           hm, 256, 0, 0, 512, 256, 512, ACT_NONE, 0, 0, 0, 1);  // += residual
    }
    k_rmsnorm<<<dim3(2), dim3(256), 0, stream>>>(hm, prm[21 + i * 41 + 40], (bf16*)0, xmf, 512, 256);
    k_take_last_concat<<<dim3((unsigned)(((long)NN * 384 + 255) / 256)), dim3(256), 0, stream>>>(
        x_orig, xmf, batch, abuf, NN);
    gemm(stream, abuf, 384, 0, wb_mrg[i], 384, 0, 1, prm[MRG0 + i * 4 + 0], 0, 0, 0,
         x_cur, 128, 0, 0, NN, 128, 384, ACT_NONE, 1,
         prm[MRG0 + i * 4 + 2], prm[MRG0 + i * 4 + 1], 1);
  }

  // ---- output GIN + pool + decoder ----
  k_zero<<<dim3(8192), dim3(256), 0, stream>>>(msum, n128);
  k_edge_scatter<<<dim3((unsigned)(((long)nE * 32 + 255) / 256)), dim3(256), 0, stream>>>(
      x_cur, esrc, edst, msum, nE);
  k_add2_bf16<<<dim3(8192), dim3(256), 0, stream>>>(x_cur, msum, bufB1, n128);
  gemm(stream, bufB1, 128, 0, wb_og, 128, 0, 1, prm[OG_B], 0, 0, 0,
       bufF1, 128, 0, 0, NN, 128, 128, ACT_NONE, 0, 0, 0, 1);
  k_pool<<<dim3(64), dim3(256), 0, stream>>>(bufF1, poolf, pool_bf);
  gemm(stream, pool_bf, 128, 0, wb_dec0, 128, 0, 1, prm[DEC0_B], 0, 0, 0,
       ddf, 128, 0, dd_bf, 128, 128, 128, ACT_RELU, 1, prm[DEC0_LNW], prm[DEC0_LNB], 1);
  gemm(stream, dd_bf, 128, 0, wb_dec1, 128, 0, 1, prm[DEC1_B], 0, 0, 0,
       (float*)d_out, 32, 0, 0, 128, 32, 128, ACT_NONE, 1, prm[DEC1_LNW], prm[DEC1_LNB], 1);
}
